// SelectiveSSM_45122926412156
// MI455X (gfx1250) — matrix operand order repaired and checked by reading, not compiled, not run on hardware
//
#include <hip/hip_runtime.h>

typedef __attribute__((ext_vector_type(16))) _Float16 v16h;
typedef __attribute__((ext_vector_type(8)))  _Float16 h8;
typedef __attribute__((ext_vector_type(8)))  float    v8f;
typedef __attribute__((ext_vector_type(4)))  float    f4;

union V16 { v16h v; h8 h[2]; };

#define D_MODEL 2048
#define D_STATE 16
#define D_INNER 4096
#define BATCH   2048

// ---------------------------------------------------------------------------
// Kernel 1: LayerNorm  x(2048x2048) -> xn (f16)
// ---------------------------------------------------------------------------
__global__ __launch_bounds__(256)
void ln_kernel(const float* __restrict__ x, const float* __restrict__ gamma,
               const float* __restrict__ beta, _Float16* __restrict__ xn) {
    const int b   = blockIdx.x;
    const int tid = threadIdx.x;
    const float* xr = x + (size_t)b * D_MODEL;

    float vals[8];
    float s = 0.f, q = 0.f;
#pragma unroll
    for (int i = 0; i < 8; ++i) {
        float v = xr[tid + i * 256];
        vals[i] = v; s += v; q += v * v;
    }
    __shared__ float sh_s[256], sh_q[256];
    sh_s[tid] = s; sh_q[tid] = q; __syncthreads();
    for (int off = 128; off > 0; off >>= 1) {
        if (tid < off) { sh_s[tid] += sh_s[tid + off]; sh_q[tid] += sh_q[tid + off]; }
        __syncthreads();
    }
    const float mu   = sh_s[0] * (1.f / D_MODEL);
    const float var  = sh_q[0] * (1.f / D_MODEL) - mu * mu;
    const float rstd = rsqrtf(var + 1e-5f);

    _Float16* xo = xn + (size_t)b * D_MODEL;
#pragma unroll
    for (int i = 0; i < 8; ++i) {
        int j = tid + i * 256;
        xo[j] = (_Float16)((vals[i] - mu) * rstd * gamma[j] + beta[j]);
    }
}

// ---------------------------------------------------------------------------
// Kernel 2/5: C(MxN) = A(MxK,f16) @ B(NxK,f32)^T  [+ residual]
// Block tile 128x128, BK=32, double-buffered LDS (one barrier per K-step).
// 8 waves in a 4(M) x 2(N) grid; wave tile 32x64 = 2x4 WMMA accumulators
// -> 8 v_wmma_f32_16x16x32_f16 per wave per K-step.
// N, K compile-time so epilogue row offsets fold into store immediates.
// LDS row stride 48 halfs (96B): 16B-aligned b128 ops, ~2-way banks.
// ---------------------------------------------------------------------------
template <int N, int K, bool ADD_RES>
__global__ __launch_bounds__(256)
void gemm_tn(const _Float16* __restrict__ A, const float* __restrict__ B,
             const float* __restrict__ R, float* __restrict__ C) {
    constexpr int LDT = 48;   // LDS tile row stride in halfs
    __shared__ __attribute__((aligned(16))) _Float16 As[2][128 * LDT];
    __shared__ __attribute__((aligned(16))) _Float16 Bs[2][128 * LDT];

    const int tid  = threadIdx.x;
    const int lane = tid & 31;
    const int w    = tid >> 5;     // 0..7
    const int wm   = w >> 1;       // 0..3  (M)
    const int wn   = w & 1;        // 0..1  (N)
    const int m0   = blockIdx.y * 128;
    const int n0   = blockIdx.x * 128;

    const int fr = lane & 15;          // fragment row within 16
    const int ka = (lane >> 4) * 8;    // A k-base (ISA 16-bit A layout)
    const int kb = (lane >> 4) * 8;    // B k-base (same k order as A)

    v8f acc[2][4] = {};

    h8 a_st[2];          // staged A regs (2 x 8 halfs)
    f4 b_st[4];          // staged B regs (f32, converted at LDS store)

    auto stage_load = [&](int kk) {
#pragma unroll
        for (int it = 0; it < 2; ++it) {
            int slot = tid + it * 256;        // 0..511
            int r    = slot >> 2;             // 0..127
            int c8   = (slot & 3) * 8;        // 0,8,16,24
            a_st[it] = *(const h8*)(A + (size_t)(m0 + r) * K + kk + c8);
            const float* bp = B + (size_t)(n0 + r) * K + kk + c8;
            b_st[2 * it]     = *(const f4*)(bp);
            b_st[2 * it + 1] = *(const f4*)(bp + 4);
        }
    };
    auto stage_store = [&](int bsel) {
#pragma unroll
        for (int it = 0; it < 2; ++it) {
            int slot = tid + it * 256;
            int r    = slot >> 2;
            int c8   = (slot & 3) * 8;
            *(h8*)(&As[bsel][r * LDT + c8]) = a_st[it];
            h8 bv;
#pragma unroll
            for (int j = 0; j < 4; ++j) {
                bv[j]     = (_Float16)b_st[2 * it][j];
                bv[j + 4] = (_Float16)b_st[2 * it + 1][j];
            }
            *(h8*)(&Bs[bsel][r * LDT + c8]) = bv;
        }
    };
    auto compute = [&](int bsel) {
        V16 af[2], bf[4];
#pragma unroll
        for (int tm = 0; tm < 2; ++tm) {
            const _Float16* p = &As[bsel][(wm * 32 + tm * 16 + fr) * LDT];
            af[tm].h[0] = *(const h8*)(p + ka);
            af[tm].h[1] = *(const h8*)(p + ka + 16);
        }
#pragma unroll
        for (int tn = 0; tn < 4; ++tn) {
            const _Float16* p = &Bs[bsel][(wn * 64 + tn * 16 + fr) * LDT];
            bf[tn].h[0] = *(const h8*)(p + kb);
            bf[tn].h[1] = *(const h8*)(p + kb + 16);
        }
#pragma unroll
        for (int tm = 0; tm < 2; ++tm)
#pragma unroll
            for (int tn = 0; tn < 4; ++tn)
                acc[tm][tn] = __builtin_amdgcn_wmma_f32_16x16x32_f16(
                    false, af[tm].v, false, bf[tn].v,
                    (short)0, acc[tm][tn], false, false);
    };

    // prologue: stage K-step 0
    stage_load(0);
    stage_store(0);
    __syncthreads();

    int buf = 0;
    for (int k0 = 32; k0 <= K; k0 += 32) {
        const bool nxt = (k0 < K);
        if (nxt) stage_load(k0);       // prefetch next tile (latency hidden)
        compute(buf);                  // 8 WMMAs from current buffer
        if (nxt) stage_store(buf ^ 1); // fill other buffer
        __syncthreads();               // single barrier per K-step
        buf ^= 1;
    }

    // epilogue: D layout -> M = v + 8*(lane>=16), Ncol = lane&15
    const int rofs = (lane >> 4) * 8;
    const int col  = lane & 15;
#pragma unroll
    for (int tm = 0; tm < 2; ++tm)
#pragma unroll
        for (int tn = 0; tn < 4; ++tn) {
            const int rbase = m0 + wm * 32 + tm * 16 + rofs;
            const int c     = n0 + wn * 64 + tn * 16 + col;
#pragma unroll
            for (int v = 0; v < 8; ++v) {
                float val = acc[tm][tn][v];
                if (ADD_RES) val += R[(size_t)(rbase + v) * N + c];
                C[(size_t)(rbase + v) * N + c] = val;
            }
        }
}

// ---------------------------------------------------------------------------
// Kernel 3: xp(2048x33) = x_main @ W_xp^T ; x_main = xz[:, :4096]
// One block per batch row; 33 register accumulators, one reduction pass.
// ---------------------------------------------------------------------------
__global__ __launch_bounds__(256)
void xp_kernel(const float* __restrict__ xz, const float* __restrict__ Wxp,
               float* __restrict__ xp) {
    const int b   = blockIdx.x;
    const int tid = threadIdx.x;
    const float* xm = xz + (size_t)b * (2 * D_INNER);

    float acc[33];
#pragma unroll
    for (int j = 0; j < 33; ++j) acc[j] = 0.f;

    for (int k = tid; k < D_INNER; k += 256) {
        const float xv = xm[k];
#pragma unroll
        for (int j = 0; j < 33; ++j) acc[j] += xv * Wxp[(size_t)j * D_INNER + k];
    }
    // wave32 shuffle reduction
#pragma unroll
    for (int j = 0; j < 33; ++j)
#pragma unroll
        for (int off = 16; off > 0; off >>= 1)
            acc[j] += __shfl_xor(acc[j], off, 32);

    __shared__ float sh[8][33];
    const int wv = tid >> 5, lane = tid & 31;
    if (lane == 0) {
#pragma unroll
        for (int j = 0; j < 33; ++j) sh[wv][j] = acc[j];
    }
    __syncthreads();
    if (tid < 33) {
        float s = 0.f;
#pragma unroll
        for (int u = 0; u < 8; ++u) s += sh[u][tid];
        xp[(size_t)b * 33 + tid] = s;
    }
}

// ---------------------------------------------------------------------------
// Kernel 4: SSM pointwise — HBM-bound (h: 512MB in + 512MB out), all b128.
// ---------------------------------------------------------------------------
__global__ __launch_bounds__(256)
void ssm_kernel(const float* __restrict__ xz, const float* __restrict__ xp,
                const float* __restrict__ log_A, const float* __restrict__ dt_bias,
                const float* __restrict__ D_param, const float* __restrict__ h,
                float* __restrict__ h_new, _Float16* __restrict__ y_gated) {
    const int b = blockIdx.x >> 4;                        // 16 blocks per row
    const int i = ((blockIdx.x & 15) << 8) + threadIdx.x; // 0..4095
    const float* xpb = xp + (size_t)b * 33;               // uniform -> s_load

    const float xm = xz[(size_t)b * (2 * D_INNER) + i];
    const float z  = xz[(size_t)b * (2 * D_INNER) + D_INNER + i];

    const float dv    = xpb[32] + dt_bias[i];
    const float delta = (dv > 20.f) ? dv : log1pf(expf(dv));   // softplus

    const size_t hb = ((size_t)b * D_INNER + i) * D_STATE;
    const f4* h4  = (const f4*)(h + hb);
    const f4* la4 = (const f4*)(log_A + (size_t)i * D_STATE);
    f4*       ho  = (f4*)(h_new + hb);

    float acc = 0.f;
#pragma unroll
    for (int qq = 0; qq < 4; ++qq) {
        f4 hv = h4[qq];
        f4 la = la4[qq];
        f4 hn;
#pragma unroll
        for (int e = 0; e < 4; ++e) {
            const int s  = qq * 4 + e;
            float Aval = expf(la[e]);
            float Abar = expf(-delta * Aval);
            hn[e] = Abar * hv[e] + delta * xpb[s] * xm;
            acc  += hn[e] * xpb[16 + s];
        }
        ho[qq] = hn;
    }
    acc += D_param[i] * xm;
    const float sig = 1.f / (1.f + expf(-z));
    y_gated[(size_t)b * D_INNER + i] = (_Float16)(acc * z * sig);
}

// ---------------------------------------------------------------------------
extern "C" void kernel_launch(void* const* d_in, const int* in_sizes, int n_in,
                              void* d_out, int out_size, void* d_ws, size_t ws_size,
                              hipStream_t stream) {
    const float* x       = (const float*)d_in[0];
    const float* h       = (const float*)d_in[1];
    const float* W_in    = (const float*)d_in[2];
    const float* W_xp    = (const float*)d_in[3];
    const float* log_A   = (const float*)d_in[4];
    const float* dt_bias = (const float*)d_in[5];
    const float* D_param = (const float*)d_in[6];
    const float* W_out   = (const float*)d_in[7];
    const float* gamma   = (const float*)d_in[8];
    const float* beta    = (const float*)d_in[9];

    float* y     = (float*)d_out;                               // 2048*2048
    float* h_new = (float*)d_out + (size_t)BATCH * D_MODEL;     // 2048*4096*16

    char* ws = (char*)d_ws;
    const size_t MB = 1024ull * 1024ull;
    _Float16* xn = (_Float16*)(ws);                  //  8 MB : 2048x2048 f16
    float*    xz = (float*)   (ws + 8 * MB);         // 64 MB : 2048x8192 f32
    float*    xp = (float*)   (ws + 72 * MB);        // <1 MB : 2048x33  f32
    _Float16* yg = (_Float16*)(ws + 73 * MB);        // 16 MB : 2048x4096 f16

    // 1) LayerNorm -> xn (f16)
    ln_kernel<<<BATCH, 256, 0, stream>>>(x, gamma, beta, xn);

    // 2) xz = xn @ W_in^T   (M=2048, N=8192, K=2048)
    gemm_tn<2 * D_INNER, D_MODEL, false>
        <<<dim3((2 * D_INNER) / 128, BATCH / 128), 256, 0, stream>>>(
            xn, W_in, nullptr, xz);

    // 3) xp = x_main @ W_xp^T  (N=33)
    xp_kernel<<<BATCH, 256, 0, stream>>>(xz, W_xp, xp);

    // 4) SSM state update (h_new into d_out) + gated output (f16)
    ssm_kernel<<<BATCH * 16, 256, 0, stream>>>(
        xz, xp, log_A, dt_bias, D_param, h, h_new, yg);

    // 5) y = y_gated @ W_out^T + x   (M=2048, N=2048, K=4096)
    gemm_tn<D_MODEL, D_INNER, true>
        <<<dim3(D_MODEL / 128, BATCH / 128), 256, 0, stream>>>(
            yg, W_out, x, y);
}
